// QueryConditionedTransportScorerV3_15934328668546
// MI455X (gfx1250) — matrix operand order repaired and checked by reading, not compiled, not run on hardware
//
#include <hip/hip_runtime.h>
#include <hip/hip_bf16.h>
#include <math.h>

typedef __attribute__((ext_vector_type(16))) _Float16 v16h;
typedef __attribute__((ext_vector_type(8)))  _Float16 v8h;
typedef __attribute__((ext_vector_type(8)))  float    v8f;

#define NQ 64
#define NW 16
#define NA 64
#define ND 512
#define NH 512

__device__ __forceinline__ float gelu_exact(float x) {
    return 0.5f * x * (1.0f + erff(x * 0.7071067811865475f));
}

// A fragment (16x32 f16, row-major tile in LDS, row stride ldk halves).
// ISA layout: lanes 0-15 -> M=lane, K in {0..7, 16..23}; lanes 16-31 -> M=lane-16, K in {8..15, 24..31}.
__device__ __forceinline__ v16h load_a_frag(const _Float16* tile, int ldk, int lane) {
    const int m  = lane & 15;
    const int kb = (lane >> 4) << 3;            // 0 or 8
    const _Float16* p = tile + m * ldk + kb;
    v8h lo = *(const v8h*)(p);                  // K = kb .. kb+7
    v8h hi = *(const v8h*)(p + 16);             // K = kb+16 .. kb+23
    return __builtin_shufflevector(lo, hi, 0,1,2,3,4,5,6,7,8,9,10,11,12,13,14,15);
}

// B fragment (32x16 f16). WT is [N][K] f16 (weight transposed): lane = column (mod 16),
// K in {kb..kb+7, kb+16..kb+23} with kb = (lane>>4)*8, matching the A fragment.
__device__ __forceinline__ v16h load_b_frag(const _Float16* __restrict__ WT, int K,
                                            int ncol0, int k0, int lane) {
    const int n  = lane & 15;
    const int kb = (lane >> 4) << 3;            // 0 or 8
    const _Float16* p = WT + (size_t)(ncol0 + n) * K + (k0 + kb);
    v8h lo = *(const v8h*)(p);
    v8h hi = *(const v8h*)(p + 16);
    return __builtin_shufflevector(lo, hi, 0,1,2,3,4,5,6,7,8,9,10,11,12,13,14,15);
}

// ---------------- prep kernels ----------------

__global__ void tcast_kernel(const float* __restrict__ W, int K, int N,
                             _Float16* __restrict__ WT) {
    int idx = blockIdx.x * blockDim.x + threadIdx.x;
    if (idx >= K * N) return;
    int k = idx / N, n = idx - k * N;
    WT[(size_t)n * K + k] = (_Float16)W[idx];
}

__global__ void pool_kernel(const float* __restrict__ qa, const float* __restrict__ qm,
                            float* __restrict__ pooled) {
    int q = blockIdx.x, t = threadIdx.x;
    float msum = 0.f;
    for (int a = 0; a < NA; a++) msum += fmaxf(qm[q * NA + a], 0.f);
    msum = fmaxf(msum, 1e-8f);
    for (int d = t; d < ND; d += blockDim.x) {
        float s = 0.f;
        for (int a = 0; a < NA; a++)
            s += fmaxf(qm[q * NA + a], 0.f) * qa[((size_t)(q * NA + a)) * ND + d];
        pooled[q * ND + d] = s / msum;
    }
}

__global__ void concat_kernel(const float* __restrict__ a, const float* __restrict__ b,
                              float* __restrict__ out) {
    int idx = blockIdx.x * blockDim.x + threadIdx.x;
    if (idx >= NW * 1024) return;
    int w = idx >> 10, d = idx & 1023;
    out[idx] = (d < 512) ? a[w * 512 + d] : b[w * 512 + (d - 512)];
}

// ---------------- generic fused MLP: LN -> GEMM -> GELU -> GEMM ----------------
// 16 rows per workgroup, 128 threads (4 waves), H = 512 output both layers.

__global__ __launch_bounds__(128) void mlp_kernel(
    const float* __restrict__ X, int Din,
    const float* __restrict__ g,  const float* __restrict__ b,
    const _Float16* __restrict__ W1T, const float* __restrict__ b1,
    const _Float16* __restrict__ W2T, const float* __restrict__ b2,
    float* __restrict__ Out) {
    __shared__ __align__(32) _Float16 Xn[16 * 1024];
    __shared__ __align__(32) _Float16 Hd[16 * NH];
    __shared__ float red[16][8][2];
    __shared__ float stats[16][2];

    const int t = threadIdx.x;
    const int row0 = blockIdx.x * 16;

    { // LN statistics
        int r = t >> 3, part = t & 7;
        const float* xr = X + (size_t)(row0 + r) * Din;
        float s1 = 0.f, s2 = 0.f;
        for (int d = part; d < Din; d += 8) { float x = xr[d]; s1 += x; s2 += x * x; }
        red[r][part][0] = s1; red[r][part][1] = s2;
    }
    __syncthreads();
    if (t < 16) {
        float s1 = 0.f, s2 = 0.f;
        for (int p = 0; p < 8; p++) { s1 += red[t][p][0]; s2 += red[t][p][1]; }
        float mean = s1 / (float)Din;
        float var  = s2 / (float)Din - mean * mean;
        stats[t][0] = mean; stats[t][1] = rsqrtf(var + 1e-5f);
    }
    __syncthreads();
    for (int idx = t; idx < 16 * Din; idx += 128) {
        int r = idx / Din, d = idx - r * Din;
        float x = X[(size_t)(row0 + r) * Din + d];
        Xn[r * Din + d] = (_Float16)((x - stats[r][0]) * stats[r][1] * g[d] + b[d]);
    }
    __syncthreads();

    const int wave = t >> 5, lane = t & 31;
    const v8f vzero = {0.f,0.f,0.f,0.f,0.f,0.f,0.f,0.f};
    const int hi8 = (lane >> 4) * 8, cl = lane & 15;

    // layer 1: [16,Din] x [Din,512]
    v8f acc1[8];
#pragma unroll
    for (int i = 0; i < 8; i++) acc1[i] = vzero;
    for (int ks = 0; ks < Din; ks += 32) {
        v16h af = load_a_frag(Xn + ks, Din, lane);
#pragma unroll
        for (int i = 0; i < 8; i++) {
            v16h bf = load_b_frag(W1T, Din, (wave * 8 + i) * 16, ks, lane);
            acc1[i] = __builtin_amdgcn_wmma_f32_16x16x32_f16(
                false, af, false, bf, (short)0, acc1[i], false, false);
        }
    }
#pragma unroll
    for (int i = 0; i < 8; i++) {
        int ncol = (wave * 8 + i) * 16 + cl;
        float bb = b1[ncol];
#pragma unroll
        for (int r = 0; r < 8; r++) {
            float y = acc1[i][r] + bb;
            Hd[(r + hi8) * NH + ncol] = (_Float16)gelu_exact(y);
        }
    }
    __syncthreads();

    // layer 2: [16,512] x [512,512]
    v8f acc2[8];
#pragma unroll
    for (int i = 0; i < 8; i++) acc2[i] = vzero;
    for (int ks = 0; ks < NH; ks += 32) {
        v16h af = load_a_frag(Hd + ks, NH, lane);
#pragma unroll
        for (int i = 0; i < 8; i++) {
            v16h bf = load_b_frag(W2T, NH, (wave * 8 + i) * 16, ks, lane);
            acc2[i] = __builtin_amdgcn_wmma_f32_16x16x32_f16(
                false, af, false, bf, (short)0, acc2[i], false, false);
        }
    }
#pragma unroll
    for (int i = 0; i < 8; i++) {
        int ncol = (wave * 8 + i) * 16 + cl;
        float bb = b2[ncol];
#pragma unroll
        for (int r = 0; r < 8; r++)
            Out[(size_t)(row0 + r + hi8) * NH + ncol] = acc2[i][r] + bb;
    }
}

// ---------------- fused relevance + softmax + entropy ----------------
// One workgroup per (q,w): 64 rows (atoms) x K=2048 fused dim -> 512 hidden -> dot rw2.

__global__ __launch_bounds__(256) void relevance_kernel(
    const float* __restrict__ AH, const float* __restrict__ QH, const float* __restrict__ CH,
    const _Float16* __restrict__ RW1T,
    const float* __restrict__ rg, const float* __restrict__ rb,
    const float* __restrict__ rb1, const float* __restrict__ rw2, const float* __restrict__ rb2,
    const float* __restrict__ class_masses,
    float* __restrict__ out_mixed, float* __restrict__ out_ent) {
    const int w = blockIdx.x, q = blockIdx.y;
    __shared__ float qv[ND], cv[ND];
    __shared__ float red[64][4][2];
    __shared__ float stats[64][2];
    __shared__ __align__(32) _Float16 Ach[64 * 256];
    __shared__ float rel[64];
    __shared__ float sm[64];
    __shared__ float scal[2];

    const int t = threadIdx.x;
    for (int d = t; d < ND; d += 256) { qv[d] = QH[q * ND + d]; cv[d] = CH[w * ND + d]; }
    if (t < 64) rel[t] = 0.f;
    __syncthreads();

    { // LN stats over the 2048-wide fused row, computed on the fly
        int a = t >> 2, part = t & 3;
        const float* ar = AH + (size_t)(w * NA + a) * ND;
        float s1 = 0.f, s2 = 0.f;
        for (int d = part; d < ND; d += 4) {
            float av = ar[d], qd = qv[d], cd = cv[d];
            float f0 = av, f1 = qd, f2 = av * qd, f3 = fabsf(av - qd) + cd;
            s1 += f0 + f1 + f2 + f3;
            s2 += f0*f0 + f1*f1 + f2*f2 + f3*f3;
        }
        red[a][part][0] = s1; red[a][part][1] = s2;
    }
    __syncthreads();
    if (t < 64) {
        float s1 = 0.f, s2 = 0.f;
        for (int p = 0; p < 4; p++) { s1 += red[t][p][0]; s2 += red[t][p][1]; }
        float mean = s1 / 2048.f;
        float var  = s2 / 2048.f - mean * mean;
        stats[t][0] = mean; stats[t][1] = rsqrtf(var + 1e-5f);
    }
    __syncthreads();

    const int wave = t >> 5, lane = t & 31;
    const v8f vzero = {0.f,0.f,0.f,0.f,0.f,0.f,0.f,0.f};
    v8f acc[4][4];
#pragma unroll
    for (int mt = 0; mt < 4; mt++)
#pragma unroll
        for (int nt = 0; nt < 4; nt++) acc[mt][nt] = vzero;

    for (int c = 0; c < 8; c++) {               // K chunks of 256
        for (int idx = t; idx < 64 * 256; idx += 256) {
            int a = idx >> 8, j = idx & 255;
            int fi = c * 256 + j, seg = fi >> 9, d = fi & 511;
            float qd = qv[d], f;
            if (seg == 1) f = qd;
            else {
                float av = AH[(size_t)(w * NA + a) * ND + d];
                if (seg == 0)      f = av;
                else if (seg == 2) f = av * qd;
                else               f = fabsf(av - qd) + cv[d];
            }
            Ach[a * 256 + j] =
                (_Float16)((f - stats[a][0]) * stats[a][1] * rg[fi] + rb[fi]);
        }
        __syncthreads();

        for (int ks = 0; ks < 256; ks += 32) {
            v16h af[4];
#pragma unroll
            for (int mt = 0; mt < 4; mt++)
                af[mt] = load_a_frag(Ach + mt * 16 * 256 + ks, 256, lane);
#pragma unroll
            for (int nt = 0; nt < 4; nt++) {
                v16h bf = load_b_frag(RW1T, 2048, (wave * 4 + nt) * 16, c * 256 + ks, lane);
#pragma unroll
                for (int mt = 0; mt < 4; mt++)
                    acc[mt][nt] = __builtin_amdgcn_wmma_f32_16x16x32_f16(
                        false, af[mt], false, bf, (short)0, acc[mt][nt], false, false);
            }
        }
        __syncthreads();
    }

    { // epilogue: +rb1, GELU, dot with rw2, reduce per-row
        const int hi8 = (lane >> 4) * 8, cl = lane & 15;
#pragma unroll
        for (int mt = 0; mt < 4; mt++) {
#pragma unroll
            for (int r = 0; r < 8; r++) {
                float ssum = 0.f;
#pragma unroll
                for (int nt = 0; nt < 4; nt++) {
                    int ncol = (wave * 4 + nt) * 16 + cl;
                    float y = acc[mt][nt][r] + rb1[ncol];
                    ssum += gelu_exact(y) * rw2[ncol];
                }
                atomicAdd(&rel[mt * 16 + r + hi8], ssum);
            }
        }
    }
    __syncthreads();

    // softmax over atoms + entropy
    if (t < 64)
        sm[t] = logf(fmaxf(class_masses[w * NA + t], 1e-8f)) + rel[t] + rb2[0];
    __syncthreads();
    if (t == 0) {
        float mx = sm[0];
        for (int a = 1; a < 64; a++) mx = fmaxf(mx, sm[a]);
        float s = 0.f;
        for (int a = 0; a < 64; a++) s += expf(sm[a] - mx);
        scal[0] = mx; scal[1] = fmaxf(s, 1e-8f);
    }
    __syncthreads();
    if (t < 64) {
        float p = expf(sm[t] - scal[0]) / scal[1];
        out_mixed[((size_t)q * NW + w) * NA + t] = p;
        sm[t] = -p * logf(fmaxf(p, 1e-8f));
    }
    __syncthreads();
    if (t == 0) {
        float e = 0.f;
        for (int a = 0; a < 64; a++) e += sm[a];
        out_ent[q * NW + w] = e;
    }
}

// ---------------- host ----------------

extern "C" void kernel_launch(void* const* d_in, const int* in_sizes, int n_in,
                              void* d_out, int out_size, void* d_ws, size_t ws_size,
                              hipStream_t stream) {
    const float* query_atoms     = (const float*)d_in[0];
    const float* query_masses    = (const float*)d_in[1];
    const float* class_atoms     = (const float*)d_in[2];
    const float* class_masses    = (const float*)d_in[3];
    const float* class_summary   = (const float*)d_in[4];
    const float* episode_context = (const float*)d_in[5];
    const float* qg  = (const float*)d_in[6],  *qb  = (const float*)d_in[7];
    const float* qw1 = (const float*)d_in[8],  *qb1 = (const float*)d_in[9];
    const float* qw2 = (const float*)d_in[10], *qb2 = (const float*)d_in[11];
    const float* ag  = (const float*)d_in[12], *ab  = (const float*)d_in[13];
    const float* aw1 = (const float*)d_in[14], *ab1 = (const float*)d_in[15];
    const float* aw2 = (const float*)d_in[16], *ab2 = (const float*)d_in[17];
    const float* cg  = (const float*)d_in[18], *cb  = (const float*)d_in[19];
    const float* cw1 = (const float*)d_in[20], *cb1 = (const float*)d_in[21];
    const float* cw2 = (const float*)d_in[22], *cb2 = (const float*)d_in[23];
    const float* rg  = (const float*)d_in[24], *rb  = (const float*)d_in[25];
    const float* rw1 = (const float*)d_in[26], *rb1 = (const float*)d_in[27];
    const float* rw2 = (const float*)d_in[28], *rb2 = (const float*)d_in[29];
    (void)in_sizes; (void)n_in; (void)out_size; (void)ws_size;

    // workspace carve-up
    char* wsp = (char*)d_ws;
    size_t off = 0;
    auto carve = [&](size_t bytes) -> char* {
        char* p = wsp + off;
        off = (off + bytes + 255) & ~(size_t)255;
        return p;
    };
    float* pooled = (float*)carve((size_t)NQ * ND * 4);
    float* Xc     = (float*)carve((size_t)NW * 1024 * 4);
    float* qh     = (float*)carve((size_t)NQ * NH * 4);
    float* ah     = (float*)carve((size_t)NW * NA * NH * 4);
    float* ch     = (float*)carve((size_t)NW * NH * 4);
    _Float16* qw1T = (_Float16*)carve((size_t)512 * 512 * 2);
    _Float16* qw2T = (_Float16*)carve((size_t)512 * 512 * 2);
    _Float16* aw1T = (_Float16*)carve((size_t)512 * 512 * 2);
    _Float16* aw2T = (_Float16*)carve((size_t)512 * 512 * 2);
    _Float16* cw1T = (_Float16*)carve((size_t)512 * 1024 * 2);
    _Float16* cw2T = (_Float16*)carve((size_t)512 * 512 * 2);
    _Float16* rw1T = (_Float16*)carve((size_t)512 * 2048 * 2);

    float* out_mixed = (float*)d_out;
    float* out_ent   = out_mixed + (size_t)NQ * NW * NA;

    // weight transpose-casts (f32 [K,N] -> f16 [N,K])
    auto tc = [&](const float* W, int K, int N, _Float16* WT) {
        int total = K * N;
        tcast_kernel<<<(total + 255) / 256, 256, 0, stream>>>(W, K, N, WT);
    };
    tc(qw1, 512, 512, qw1T);   tc(qw2, 512, 512, qw2T);
    tc(aw1, 512, 512, aw1T);   tc(aw2, 512, 512, aw2T);
    tc(cw1, 1024, 512, cw1T);  tc(cw2, 512, 512, cw2T);
    tc(rw1, 2048, 512, rw1T);

    pool_kernel<<<NQ, 256, 0, stream>>>(query_atoms, query_masses, pooled);
    concat_kernel<<<(NW * 1024 + 255) / 256, 256, 0, stream>>>(class_summary, episode_context, Xc);

    // qh: [64,512] ; ah: [1024,512] ; ch: [16,512]
    mlp_kernel<<<NQ / 16, 128, 0, stream>>>(pooled, 512, qg, qb, qw1T, qb1, qw2T, qb2, qh);
    mlp_kernel<<<(NW * NA) / 16, 128, 0, stream>>>(class_atoms, 512, ag, ab, aw1T, ab1, aw2T, ab2, ah);
    mlp_kernel<<<NW / 16, 128, 0, stream>>>(Xc, 1024, cg, cb, cw1T, cb1, cw2T, cb2, ch);

    relevance_kernel<<<dim3(NW, NQ), 256, 0, stream>>>(
        ah, qh, ch, rw1T, rg, rb, rb1, rw2, rb2, class_masses, out_mixed, out_ent);
}
